// EncoderG_36051955482709
// MI455X (gfx1250) — matrix operand order repaired and checked by reading, not compiled, not run on hardware
//
#include <hip/hip_runtime.h>

typedef __bf16 bf16;
typedef __attribute__((ext_vector_type(16))) __bf16 v16bf;
typedef __attribute__((ext_vector_type(8)))  __bf16 v8bf;
typedef __attribute__((ext_vector_type(8)))  float  v8f;
typedef __attribute__((ext_vector_type(4)))  int    v4i;

static constexpr int NN = 4096;  // nodes
static constexpr int DD = 512;   // in dim
static constexpr int HH = 256;   // hidden
static constexpr int ZZ = 128;   // latent
static constexpr float EPS = 1e-3f;

static constexpr int BM = 128;
static constexpr int BN = 128;
static constexpr int BK = 32;
static constexpr int LPAD = 8;
static constexpr int NT = 128;   // threads per GEMM block (4 waves, each 64x64)

#if defined(__has_builtin)
#if __has_builtin(__builtin_amdgcn_global_load_async_to_lds_b128) && \
    __has_builtin(__builtin_amdgcn_s_wait_asynccnt)
#define USE_ASYNC_LDS 1
#endif
#endif

#ifdef USE_ASYNC_LDS
typedef __attribute__((address_space(1))) v4i gv4i;   // global int4*
typedef __attribute__((address_space(3))) v4i lv4i;   // LDS int4*
#endif

// C(MxN) = A(MxK,bf16 row-major) @ BT(NxK,bf16 row-major == B col-major) [+bias][+C]
// Outputs: Cf (f32, optional accumulate) and/or Cb (bf16 row-major) and/or CbT (bf16 transposed).
// 4 waves, each computing a 64x64 micro-tile = 4x4 WMMA fragments -> 16 v_wmma / k-step.
__global__ __launch_bounds__(NT) void gemm_bf16_wmma(
    const bf16* __restrict__ A, int lda,
    const bf16* __restrict__ BT, int ldbt,
    float* __restrict__ Cf, int ldc,
    bf16* __restrict__ Cb, int ldcb,
    bf16* __restrict__ CbT, int ldct,
    const float* __restrict__ bias,
    int accum, int K)
{
  __shared__ bf16 As[2][BM][BK + LPAD];   // A tile (row-major)
  __shared__ bf16 Bs[2][BN][BK + LPAD];   // B tile, [n][k] (from BT, no transpose needed)

  const int t    = threadIdx.x;
  const int lane = t & 31;
  const int wave = t >> 5;
  const int lr   = lane & 15;
  const int half = lane >> 4;
  const int waveM = wave & 1;   // 2 waves along M (64 rows each)
  const int waveN = wave >> 1;  // 2 waves along N (64 cols each)
  const int m0 = blockIdx.y * BM;
  const int n0 = blockIdx.x * BN;

  // staging coords: 128x32 tile = 512 x 16B chunks, 4 per thread
  int rowi[4], coli[4];
#pragma unroll
  for (int c = 0; c < 4; ++c) {
    int id = t + c * NT;
    rowi[c] = id >> 2;
    coli[c] = (id & 3) << 3;
  }

  v8f acc[4][4];
#pragma unroll
  for (int i = 0; i < 4; ++i)
#pragma unroll
    for (int j = 0; j < 4; ++j) acc[i][j] = {};

  // ---- prologue: stage tile k0=0 into buffer 0 ----
#ifdef USE_ASYNC_LDS
#pragma unroll
  for (int c = 0; c < 4; ++c) {
    __builtin_amdgcn_global_load_async_to_lds_b128(
        (gv4i*)(A + (size_t)(m0 + rowi[c]) * lda + coli[c]),
        (lv4i*)&As[0][rowi[c]][coli[c]], 0, 0);
    __builtin_amdgcn_global_load_async_to_lds_b128(
        (gv4i*)(BT + (size_t)(n0 + rowi[c]) * ldbt + coli[c]),
        (lv4i*)&Bs[0][rowi[c]][coli[c]], 0, 0);
  }
  __builtin_amdgcn_s_wait_asynccnt(0);
#else
#pragma unroll
  for (int c = 0; c < 4; ++c) {
    v8bf a = *(const v8bf*)(A  + (size_t)(m0 + rowi[c]) * lda  + coli[c]);
    v8bf b = *(const v8bf*)(BT + (size_t)(n0 + rowi[c]) * ldbt + coli[c]);
    *(v8bf*)&As[0][rowi[c]][coli[c]] = a;
    *(v8bf*)&Bs[0][rowi[c]][coli[c]] = b;
  }
#endif
  __syncthreads();

  int buf = 0;
  for (int k0 = 0; k0 < K; k0 += BK) {
    const bool has_next = (k0 + BK) < K;
    const int kn = k0 + BK;

    // ---- issue next tile's loads (overlap with WMMA below) ----
#ifdef USE_ASYNC_LDS
    if (has_next) {
#pragma unroll
      for (int c = 0; c < 4; ++c) {
        __builtin_amdgcn_global_load_async_to_lds_b128(
            (gv4i*)(A + (size_t)(m0 + rowi[c]) * lda + kn + coli[c]),
            (lv4i*)&As[buf ^ 1][rowi[c]][coli[c]], 0, 0);
        __builtin_amdgcn_global_load_async_to_lds_b128(
            (gv4i*)(BT + (size_t)(n0 + rowi[c]) * ldbt + kn + coli[c]),
            (lv4i*)&Bs[buf ^ 1][rowi[c]][coli[c]], 0, 0);
      }
    }
#else
    v8bf pa[4], pb[4];
    if (has_next) {
#pragma unroll
      for (int c = 0; c < 4; ++c) {
        pa[c] = *(const v8bf*)(A  + (size_t)(m0 + rowi[c]) * lda  + kn + coli[c]);
        pb[c] = *(const v8bf*)(BT + (size_t)(n0 + rowi[c]) * ldbt + kn + coli[c]);
      }
    }
#endif

    // ---- fragments per ISA 16-bit A/B lane layouts ----
    v16bf afrag[4];
#pragma unroll
    for (int fm = 0; fm < 4; ++fm) {
      int row = waveM * 64 + fm * 16 + lr;
      v8bf lo = *(const v8bf*)&As[buf][row][half * 8];        // K = h8+0..7
      v8bf hi = *(const v8bf*)&As[buf][row][16 + half * 8];   // K = 16+h8+0..7
      afrag[fm] = __builtin_shufflevector(lo, hi, 0,1,2,3,4,5,6,7,8,9,10,11,12,13,14,15);
    }
    v16bf bfrag[4];
#pragma unroll
    for (int fn = 0; fn < 4; ++fn) {
      int coln = waveN * 64 + fn * 16 + lr;                   // column N = lr
      v8bf lo = *(const v8bf*)&Bs[buf][coln][half * 8];       // K = h8+0..7
      v8bf hi = *(const v8bf*)&Bs[buf][coln][16 + half * 8];  // K = 16+h8+0..7
      bfrag[fn] = __builtin_shufflevector(lo, hi, 0,1,2,3,4,5,6,7,8,9,10,11,12,13,14,15);
    }

#pragma unroll
    for (int fm = 0; fm < 4; ++fm)
#pragma unroll
      for (int fn = 0; fn < 4; ++fn)
        acc[fm][fn] = __builtin_amdgcn_wmma_f32_16x16x32_bf16(
            false, afrag[fm], false, bfrag[fn], (short)0, acc[fm][fn], false, false);

    // ---- commit next tile, single barrier per iteration ----
#ifdef USE_ASYNC_LDS
    if (has_next) __builtin_amdgcn_s_wait_asynccnt(0);
#else
    if (has_next) {
#pragma unroll
      for (int c = 0; c < 4; ++c) {
        *(v8bf*)&As[buf ^ 1][rowi[c]][coli[c]] = pa[c];
        *(v8bf*)&Bs[buf ^ 1][rowi[c]][coli[c]] = pb[c];
      }
    }
#endif
    __syncthreads();
    buf ^= 1;
  }

  // ---- epilogue: C layout — VGPR r -> M = half*8 + r, N = lane&15 ----
#pragma unroll
  for (int fm = 0; fm < 4; ++fm) {
#pragma unroll
    for (int fn = 0; fn < 4; ++fn) {
      int col  = n0 + waveN * 64 + fn * 16 + lr;
      int rowb = m0 + waveM * 64 + fm * 16 + half * 8;
      float bv = bias ? bias[col] : 0.0f;
      float vals[8];
#pragma unroll
      for (int r = 0; r < 8; ++r) vals[r] = acc[fm][fn][r] + bv;
      if (Cb) {
#pragma unroll
        for (int r = 0; r < 8; ++r)
          Cb[(size_t)(rowb + r) * ldcb + col] = (bf16)vals[r];
      }
      if (CbT) {  // transposed copy: contiguous along M -> one b128 store
        v8bf tv;
#pragma unroll
        for (int r = 0; r < 8; ++r) tv[r] = (bf16)vals[r];
        *(v8bf*)(CbT + (size_t)col * ldct + rowb) = tv;
      }
      if (Cf) {
#pragma unroll
        for (int r = 0; r < 8; ++r) {
          size_t o = (size_t)(rowb + r) * ldc + col;
          float v = vals[r];
          if (accum) v += Cf[o];
          Cf[o] = v;
        }
      }
    }
  }
}

// ---------------- helpers ----------------
__global__ void cvt_f32_bf16(const float* __restrict__ s, bf16* __restrict__ d, long n) {
  long i = (long)blockIdx.x * blockDim.x + threadIdx.x;
  long st = (long)gridDim.x * blockDim.x;
  for (; i < n; i += st) d[i] = (bf16)s[i];
}

// d (C x R) = transpose of s (R x C), f32 -> bf16
__global__ void cvt_f32_bf16_T(const float* __restrict__ s, int R, int C,
                               bf16* __restrict__ d, long n /* R*C */) {
  long i = (long)blockIdx.x * blockDim.x + threadIdx.x;
  long st = (long)gridDim.x * blockDim.x;
  for (; i < n; i += st) {
    long c = i / R;
    int  r = (int)(i - c * R);
    d[i] = (bf16)s[(size_t)r * C + c];
  }
}

__global__ void cvt_f32_bf16_strided(const float* __restrict__ s, int cols,
                                     bf16* __restrict__ d, int ldd, long n) {
  long i = (long)blockIdx.x * blockDim.x + threadIdx.x;
  long st = (long)gridDim.x * blockDim.x;
  for (; i < n; i += st) {
    long row = i / cols;
    int  c   = (int)(i - row * cols);
    d[row * (long)ldd + c] = (bf16)s[i];
  }
}

__global__ void bn_relu_to_bf16(const float* __restrict__ h,
                                const float* __restrict__ gamma,
                                const float* __restrict__ beta,
                                const float* __restrict__ mean,
                                const float* __restrict__ var,
                                bf16* __restrict__ hbf, int ldb,
                                bf16* __restrict__ hT, int ldt,
                                int Hdim, long n) {
  long i = (long)blockIdx.x * blockDim.x + threadIdx.x;
  long st = (long)gridDim.x * blockDim.x;
  for (; i < n; i += st) {
    long row = i / Hdim;
    int  c   = (int)(i - row * Hdim);
    float v = (h[i] - mean[c]) * rsqrtf(var[c] + EPS) * gamma[c] + beta[c];
    v = fmaxf(v, 0.0f);
    bf16 bv = (bf16)v;
    hbf[row * (long)ldb + c] = bv;
    hT[(size_t)c * ldt + row] = bv;
  }
}

__global__ void zero_f32(float* __restrict__ p, long n) {
  long i = (long)blockIdx.x * blockDim.x + threadIdx.x;
  long st = (long)gridDim.x * blockDim.x;
  for (; i < n; i += st) p[i] = 0.0f;
}

__global__ void add_f32(float* __restrict__ dst, const float* __restrict__ src, long n) {
  long i = (long)blockIdx.x * blockDim.x + threadIdx.x;
  long st = (long)gridDim.x * blockDim.x;
  for (; i < n; i += st) dst[i] += src[i];
}

// ---------------- host orchestration ----------------
extern "C" void kernel_launch(void* const* d_in, const int* in_sizes, int n_in,
                              void* d_out, int out_size, void* d_ws, size_t ws_size,
                              hipStream_t stream) {
  (void)in_sizes; (void)n_in; (void)ws_size;

  const float* x = (const float*)d_in[0];
  const float* Ain[2] = { (const float*)d_in[1], (const float*)d_in[2] };

  char* ws = (char*)d_ws;
  size_t off = 0;
  auto carve = [&](size_t bytes) {
    char* p = ws + off;
    off += (bytes + 255) & ~(size_t)255;
    return p;
  };
  bf16*  Abf     = (bf16*)carve((size_t)NN * NN * sizeof(bf16));       // 32 MB
  bf16*  concat1 = (bf16*)carve((size_t)NN * 4 * DD * sizeof(bf16));   // 16 MB
  bf16*  concat2 = (bf16*)carve((size_t)NN * 4 * HH * sizeof(bf16));   //  8 MB
  bf16*  XT      = (bf16*)carve((size_t)DD * NN * sizeof(bf16));       //  4 MB (x transposed)
  bf16*  BTa     = (bf16*)carve((size_t)DD * NN * sizeof(bf16));       //  4 MB ping
  bf16*  BTb     = (bf16*)carve((size_t)DD * NN * sizeof(bf16));       //  4 MB pong
  float* hbuf    = (float*)carve((size_t)NN * HH * sizeof(float));     //  4 MB
  float* zbuf    = (float*)carve((size_t)NN * ZZ * sizeof(float));     //  2 MB
  bf16*  W1T     = (bf16*)carve((size_t)4 * DD * HH * sizeof(bf16));   // 256 x 2048
  bf16*  W2T     = (bf16*)carve((size_t)4 * HH * ZZ * sizeof(bf16));   // 128 x 1024
  bf16*  WmT     = (bf16*)carve((size_t)HH * ZZ * sizeof(bf16));       // 128 x 256

  const dim3 blk(256);
  const dim3 gblk(NT);

  zero_f32<<<dim3(256), blk, 0, stream>>>((float*)d_out, (long)out_size);

  // x -> bf16 row-major into concat1 slice 0, and transposed into XT (both once)
  cvt_f32_bf16_strided<<<dim3(2048), blk, 0, stream>>>(x, DD, concat1, 4 * DD, (long)NN * DD);
  cvt_f32_bf16_T<<<dim3(2048), blk, 0, stream>>>(x, NN, DD, XT, (long)NN * DD);

  for (int b = 0; b < 2; ++b) {
    const int base = 3 + 10 * b;
    const float* W1 = (const float*)d_in[base + 0];
    const float* b1 = (const float*)d_in[base + 1];
    const float* W2 = (const float*)d_in[base + 2];
    const float* b2 = (const float*)d_in[base + 3];
    const float* Wm = (const float*)d_in[base + 4];
    const float* bm = (const float*)d_in[base + 5];
    const float* ga = (const float*)d_in[base + 6];
    const float* be = (const float*)d_in[base + 7];
    const float* mu = (const float*)d_in[base + 8];
    const float* va = (const float*)d_in[base + 9];

    cvt_f32_bf16<<<dim3(4096), blk, 0, stream>>>(Ain[b], Abf, (long)NN * NN);
    cvt_f32_bf16_T<<<dim3(512), blk, 0, stream>>>(W1, 4 * DD, HH, W1T, (long)4 * DD * HH);
    cvt_f32_bf16_T<<<dim3(256), blk, 0, stream>>>(W2, 4 * HH, ZZ, W2T, (long)4 * HH * ZZ);
    cvt_f32_bf16_T<<<dim3(64),  blk, 0, stream>>>(Wm, HH, ZZ, WmT, (long)HH * ZZ);

    // --- TAGConv 1 hops: out = A @ hop_prev; row-major into concat1 slice + transposed ping-pong ---
    const bf16* hopBT[3] = { XT, BTa, BTb };
    bf16*       hopTo[3] = { BTa, BTb, nullptr };
    for (int h = 1; h <= 3; ++h) {
      gemm_bf16_wmma<<<dim3(DD / BN, NN / BM), gblk, 0, stream>>>(
          Abf, NN, hopBT[h - 1], NN,
          nullptr, 0, concat1 + (size_t)h * DD, 4 * DD, hopTo[h - 1], NN, nullptr, 0, NN);
    }
    // --- dense 1: hbuf = concat1 @ W1 + b1 ---
    gemm_bf16_wmma<<<dim3(HH / BN, NN / BM), gblk, 0, stream>>>(
        concat1, 4 * DD, W1T, 4 * DD, hbuf, HH, nullptr, 0, nullptr, 0, b1, 0, 4 * DD);
    // --- BN + ReLU -> bf16 h into concat2 slice 0 (row-major) and BTa (transposed) ---
    bn_relu_to_bf16<<<dim3(1024), blk, 0, stream>>>(hbuf, ga, be, mu, va,
                                                    concat2, 4 * HH, BTa, NN, HH, (long)NN * HH);
    // --- TAGConv 2 hops on h ---
    const bf16* hopBT2[3] = { BTa, BTb, BTa };
    bf16*       hopTo2[3] = { BTb, BTa, nullptr };
    for (int h = 1; h <= 3; ++h) {
      gemm_bf16_wmma<<<dim3(HH / BN, NN / BM), gblk, 0, stream>>>(
          Abf, NN, hopBT2[h - 1], NN,
          nullptr, 0, concat2 + (size_t)h * HH, 4 * HH, hopTo2[h - 1], NN, nullptr, 0, NN);
    }
    // --- dense 2: zbuf = concat2 @ W2 + b2 ---
    gemm_bf16_wmma<<<dim3(ZZ / BN, NN / BM), gblk, 0, stream>>>(
        concat2, 4 * HH, W2T, 4 * HH, zbuf, ZZ, nullptr, 0, nullptr, 0, b2, 0, 4 * HH);
    // --- residual: zbuf += h @ Wm + bm ---
    gemm_bf16_wmma<<<dim3(ZZ / BN, NN / BM), gblk, 0, stream>>>(
        concat2, 4 * HH, WmT, HH, zbuf, ZZ, nullptr, 0, nullptr, 0, bm, 1, HH);
    // --- accumulate branch into output ---
    add_f32<<<dim3(512), blk, 0, stream>>>((float*)d_out, zbuf, (long)NN * ZZ);
  }
}
